// Model_12532714569815
// MI455X (gfx1250) — matrix operand order repaired and checked by reading, not compiled, not run on hardware
//
#include <hip/hip_runtime.h>

// LSTM: B=256, T=256, H=512. Sequential over T -> 256 dependent step kernels.
// Recurrent GEMM per step via v_wmma_f32_16x16x32_bf16. K-loop is software-
// pipelined (prefetch distance 2, enforced with sched_barrier) and uses
// immediate-offset addressing so the hot loop has zero address VALU.
#define B_  256
#define T_  256
#define H_  512
#define G4H (4 * H_)

typedef __attribute__((ext_vector_type(8)))  __bf16 bf16x8;
typedef __attribute__((ext_vector_type(16))) __bf16 bf16x16;
typedef __attribute__((ext_vector_type(8)))  float  f32x8;

__device__ __forceinline__ unsigned short f2bf(float f) {
    unsigned int u = __float_as_uint(f);
    u += 0x7FFFu + ((u >> 16) & 1u);   // round-to-nearest-even
    return (unsigned short)(u >> 16);
}

__device__ __forceinline__ float sigm(float x) {
    return 1.0f / (1.0f + __expf(-x));
}

__device__ __forceinline__ float tanh_fast(float x) {
    float e = __expf(-2.0f * x);
    return (1.0f - e) / (1.0f + e);
}

__device__ __forceinline__ bf16x16 mk16(uint4 lo, uint4 hi) {
    bf16x8 l = __builtin_bit_cast(bf16x8, lo);
    bf16x8 h = __builtin_bit_cast(bf16x8, hi);
    return __builtin_shufflevector(l, h,
               0, 1, 2, 3, 4, 5, 6, 7, 8, 9, 10, 11, 12, 13, 14, 15);
}

// One-time prep: W_hh fp32 -> bf16 (2048x512, row-major over K), bias = b_ih + b_hh.
__global__ void prep_kernel(const float* __restrict__ Whh,
                            const float* __restrict__ bih,
                            const float* __restrict__ bhh,
                            unsigned short* __restrict__ Wbf,
                            float* __restrict__ bias) {
    unsigned i = blockIdx.x * blockDim.x + threadIdx.x;
    if (i < G4H * H_) Wbf[i] = f2bf(Whh[i]);
    if (i < G4H)      bias[i] = bih[i] + bhh[i];
}

// Per-call init: h0 = 0 (bf16), c = 0, out[b,t] = b_out (atomics add partial y later).
__global__ void init_kernel(const float* __restrict__ bout,
                            unsigned short* __restrict__ h0,
                            float* __restrict__ cst,
                            float* __restrict__ out) {
    unsigned i = blockIdx.x * blockDim.x + threadIdx.x;
    if (i < B_ * H_) { h0[i] = 0; cst[i] = 0.0f; }
    if (i < B_ * T_) out[i] = bout[0];
}

// One LSTM time step.
// Grid: (H/128, B/16). Block: 256 threads = 8 waves. Wave w owns 16 rows x 16 h-cols
// and computes all 4 gate tiles (i,f,g,o) for those columns -> wave-local cell update.
__global__ __launch_bounds__(256, 1) void lstm_step(
    const float* __restrict__ input,          // [B, T]
    const float* __restrict__ Wih,            // [4H] (input_size == 1)
    const float* __restrict__ Wout,           // [H]
    const unsigned short* __restrict__ Wbf,   // [4H, H] bf16
    const float* __restrict__ bias,           // [4H]
    const unsigned short* __restrict__ hin,   // [B, H] bf16 (state at t)
    unsigned short* __restrict__ hout,        // [B, H] bf16 (state at t+1)
    float* __restrict__ cst,                  // [B, H] f32 cell state
    float* __restrict__ out,                  // [B, T]
    int t) {
    // 16 x 512 bf16 h-tile; stride 520 (1040B) avoids 64-bank conflicts on b128 reads.
    __shared__ __align__(16) unsigned short hs[16][520];
    __shared__ float xs[16];

    const int tid     = threadIdx.x;
    const int rowbase = blockIdx.y * 16;   // batch rows
    const int colblk  = blockIdx.x * 128;  // h columns handled by this block

    // Stage h rows [rowbase, rowbase+16) x all 512 cols into LDS (uint4 = 8 bf16).
#pragma unroll
    for (int i = 0; i < 4; ++i) {
        int idx = tid + i * 256;           // 1024 uint4 chunks total
        int r   = idx >> 6;                // / 64
        int c8  = (idx & 63) << 3;         // * 8 elements
        uint4 v = *(const uint4*)(hin + (unsigned)((rowbase + r) * H_ + c8));
        *(uint4*)&hs[r][c8] = v;
    }
    if (tid < 16) xs[tid] = input[(unsigned)((rowbase + tid) * T_ + t)];
    __syncthreads();

    const int wave = tid >> 5;
    const int lane = tid & 31;
    const int n    = lane & 15;            // tile column / A row (M) for this lane
    const int hf   = lane >> 4;            // lane half
    const int j    = colblk + wave * 16 + n;  // global h column of this lane's D entries

    f32x8 acc[4];
#pragma unroll
    for (int g = 0; g < 4; ++g)
#pragma unroll
        for (int e = 0; e < 8; ++e) acc[g][e] = 0.0f;

    const int abase = hf * 8;    // A frag: lanes 0-15 hold K 0-7/16-23; 16-31 hold 8-15/24-31
    const int kbB   = hf * 8;    // B frag: lanes 0-15 hold K 0-7/16-23; 16-31 hold 8-15/24-31

    // Per-gate base pointers; all 16 K-chunks then use immediate offsets only.
    const uint4* pB[4];
#pragma unroll
    for (int g = 0; g < 4; ++g)
        pB[g] = (const uint4*)(Wbf + (unsigned)(g * (H_ * H_) + j * H_ + kbB));
    const uint4* pA = (const uint4*)&hs[n][abase];   // chunk c: pA[4c], pA[4c+2]

    // Rotating fragment buffers, prefetch distance 2.
    uint4 aL[3], aH[3];
    uint4 bL[3][4], bH[3][4];

    auto loadA = [&](int c, int s) {
        aL[s] = pA[4 * c];
        aH[s] = pA[4 * c + 2];
    };
    auto loadB = [&](int c, int s) {
#pragma unroll
        for (int g = 0; g < 4; ++g) {
            bL[s][g] = pB[g][4 * c];
            bH[s][g] = pB[g][4 * c + 2];
        }
    };
    auto domma = [&](int s) {
        bf16x16 a = mk16(aL[s], aH[s]);
#pragma unroll
        for (int g = 0; g < 4; ++g) {
            bf16x16 b = mk16(bL[s][g], bH[s][g]);
            acc[g] = __builtin_amdgcn_wmma_f32_16x16x32_bf16(
                         false, a, false, b, (short)0, acc[g], false, false);
        }
    };

    loadA(0, 0); loadB(0, 0);
    loadA(1, 1); loadB(1, 1);
#pragma unroll
    for (int c = 0; c < 16; ++c) {
        const int cur = c % 3;
        const int pre = (c + 2) % 3;
        if (c + 2 < 16) {               // keep ~16-20 loads in flight
            loadA(c + 2, pre);
            loadB(c + 2, pre);
        }
        __builtin_amdgcn_sched_barrier(0);   // loads stay ahead of this chunk's WMMAs
        domma(cur);
    }

    // Epilogue: per-lane column constants, then cell update per row.
    const float wih0 = Wih[j],          wih1 = Wih[H_ + j];
    const float wih2 = Wih[2 * H_ + j], wih3 = Wih[3 * H_ + j];
    const float bs0 = bias[j],          bs1 = bias[H_ + j];
    const float bs2 = bias[2 * H_ + j], bs3 = bias[3 * H_ + j];
    const float wo  = Wout[j];

    float yp[8];
#pragma unroll
    for (int r = 0; r < 8; ++r) {
        // D layout: M = r + 8*hf, N = n
        const int   brow = rowbase + hf * 8 + r;
        const float x    = xs[hf * 8 + r];
        float gi = acc[0][r] + x * wih0 + bs0;
        float gf = acc[1][r] + x * wih1 + bs1;
        float gg = acc[2][r] + x * wih2 + bs2;
        float go = acc[3][r] + x * wih3 + bs3;
        const unsigned ci = (unsigned)(brow * H_ + j);
        float cn = sigm(gf) * cst[ci] + sigm(gi) * tanh_fast(gg);
        float hn = sigm(go) * tanh_fast(cn);
        cst[ci]  = cn;
        hout[ci] = f2bf(hn);
        yp[r]    = hn * wo;
    }

    // y partials: reduce over 16 lanes of each half, atomic-add into out[b, t].
#pragma unroll
    for (int r = 0; r < 8; ++r) {
        float v = yp[r];
        v += __shfl_xor(v, 1, 32);
        v += __shfl_xor(v, 2, 32);
        v += __shfl_xor(v, 4, 32);
        v += __shfl_xor(v, 8, 32);
        if (n == 0) {
            const int brow = rowbase + hf * 8 + r;
            atomicAdd(&out[(unsigned)(brow * T_ + t)], v);
        }
    }
}

extern "C" void kernel_launch(void* const* d_in, const int* in_sizes, int n_in,
                              void* d_out, int out_size, void* d_ws, size_t ws_size,
                              hipStream_t stream) {
    const float* input = (const float*)d_in[0];  // [B, T]
    const float* Wih   = (const float*)d_in[1];  // [4H, 1]
    const float* Whh   = (const float*)d_in[2];  // [4H, H]
    const float* bih   = (const float*)d_in[3];  // [4H]
    const float* bhh   = (const float*)d_in[4];  // [4H]
    const float* Wout  = (const float*)d_in[5];  // [1, H]
    const float* bout  = (const float*)d_in[6];  // [1]
    float* out = (float*)d_out;                  // [B, T]

    // Workspace layout (all 16B aligned):
    char* ws = (char*)d_ws;
    unsigned short* Wbf  = (unsigned short*)(ws);              // 2048*512*2 = 2097152
    float*          bias = (float*)(ws + 2097152);             // 2048*4     =    8192
    unsigned short* h0   = (unsigned short*)(ws + 2105344);    // 256*512*2  =  262144
    unsigned short* h1   = (unsigned short*)(ws + 2367488);    // 256*512*2  =  262144
    float*          cst  = (float*)(ws + 2629632);             // 256*512*4  =  524288

    prep_kernel<<<(G4H * H_ + 255) / 256, 256, 0, stream>>>(Whh, bih, bhh, Wbf, bias);
    init_kernel<<<(B_ * H_ + 255) / 256, 256, 0, stream>>>(bout, h0, cst, out);

    dim3 grid(H_ / 128, B_ / 16);  // (4, 16)
    for (int t = 0; t < T_; ++t) {
        const unsigned short* hin = (t & 1) ? h1 : h0;
        unsigned short*      hnew = (t & 1) ? h0 : h1;
        lstm_step<<<grid, 256, 0, stream>>>(input, Wih, Wout, Wbf, bias,
                                            hin, hnew, cst, out, t);
    }
}
